// FasterRCNN_5007931867216
// MI455X (gfx1250) — matrix operand order repaired and checked by reading, not compiled, not run on hardware
//
#include <hip/hip_runtime.h>

typedef __bf16 bf16;
typedef __attribute__((ext_vector_type(16))) __bf16 v16bf;
typedef __attribute__((ext_vector_type(8)))  float  v8f;
typedef __attribute__((ext_vector_type(4)))  float  v4f;
typedef unsigned long long u64;

#define N_ANCHORS 22500
#define SORT_N    32768
#define N_PRE     6000
#define N_POST    300
#define R_SAMPLE  128
#define IMG_SZ    800.0f
#define MIN_SZ    16.0f
#define NMS_TH    0.7f
#define NSUB      4          // 4 N-subtiles of 16 -> 64 columns per workgroup

__device__ __forceinline__ v8f wmma_bf16(v16bf a, v16bf b, v8f c) {
    return __builtin_amdgcn_wmma_f32_16x16x32_bf16(false, a, false, b, (short)0, c,
                                                   false, false);
}

// ---------------------------------------------------------------------------
// Stage A1: decode anchors+deltas -> clipped rois; fg score; packed sort keys.
// key = (scoreBits << 32) | ~idx  (sigmoid score > 0; invalid -> hi32 = 0;
// padding idx>=22500 -> key=0).  Descending sort == jax top_k w/ stable ties.
// ---------------------------------------------------------------------------
__global__ void k_props(const float* __restrict__ logits,
                        const float* __restrict__ deltas,
                        const float* __restrict__ anchors,
                        float* __restrict__ rois, u64* __restrict__ keys) {
    int i = blockIdx.x * 256 + threadIdx.x;
    if (i >= SORT_N) return;
    if (i >= N_ANCHORS) { keys[i] = 0ull; return; }
    float a0 = anchors[4*i+0], a1 = anchors[4*i+1];
    float a2 = anchors[4*i+2], a3 = anchors[4*i+3];
    float h = a2 - a0, w = a3 - a1;
    float cy = a0 + 0.5f * h, cx = a1 + 0.5f * w;
    float d0 = deltas[4*i+0], d1 = deltas[4*i+1];
    float d2 = deltas[4*i+2], d3 = deltas[4*i+3];
    float ncy = d0 * h + cy, ncx = d1 * w + cx;
    float nh = expf(d2) * h, nw = expf(d3) * w;
    float y1 = fminf(fmaxf(ncy - 0.5f * nh, 0.f), IMG_SZ);
    float x1 = fminf(fmaxf(ncx - 0.5f * nw, 0.f), IMG_SZ);
    float y2 = fminf(fmaxf(ncy + 0.5f * nh, 0.f), IMG_SZ);
    float x2 = fminf(fmaxf(ncx + 0.5f * nw, 0.f), IMG_SZ);
    rois[4*i+0] = y1; rois[4*i+1] = x1; rois[4*i+2] = y2; rois[4*i+3] = x2;
    bool valid = ((y2 - y1) >= MIN_SZ) && ((x2 - x1) >= MIN_SZ);
    float score = 1.0f / (1.0f + expf(logits[2*i] - logits[2*i+1]));
    unsigned sb = valid ? __float_as_uint(score) : 0u;
    keys[i] = ((u64)sb << 32) | (u64)(0xFFFFFFFFu - (unsigned)i);
}

// Global bitonic sort step (descending).
__global__ void k_bitonic(u64* __restrict__ keys, int j, int k) {
    int i = blockIdx.x * 256 + threadIdx.x;
    int ixj = i ^ j;
    if (ixj > i && i < SORT_N) {
        u64 a = keys[i], b = keys[ixj];
        bool up = ((i & k) == 0);               // descending overall
        if (up ? (a < b) : (a > b)) { keys[i] = b; keys[ixj] = a; }
    }
}

// Gather top-6000 boxes in score order.
__global__ void k_gather(const u64* __restrict__ keys,
                         const float* __restrict__ rois,
                         float* __restrict__ bsorted) {
    int q = blockIdx.x * 256 + threadIdx.x;
    if (q >= N_PRE) return;
    unsigned ord = 0xFFFFFFFFu - (unsigned)(keys[q] & 0xFFFFFFFFull);
    if (ord < N_ANCHORS) {
        bsorted[4*q+0] = rois[4*ord+0]; bsorted[4*q+1] = rois[4*ord+1];
        bsorted[4*q+2] = rois[4*ord+2]; bsorted[4*q+3] = rois[4*ord+3];
    } else {
        bsorted[4*q+0] = 0.f; bsorted[4*q+1] = 0.f;
        bsorted[4*q+2] = 0.f; bsorted[4*q+3] = 0.f;
    }
}

// Greedy NMS on sorted scores: pick = first live index (== argmax); -inf
// entries (hi32==0) start dead; exhausted -> index 0 (matches jnp.argmax).
__global__ void k_nms(const float* __restrict__ boxes,
                      const u64* __restrict__ keys,
                      float* __restrict__ rois_final) {
    __shared__ int s_cur;
    __shared__ unsigned char dead[N_PRE];
    int tid = threadIdx.x;
    for (int q = tid; q < N_PRE; q += blockDim.x)
        dead[q] = ((unsigned)(keys[q] >> 32) == 0u) ? 1 : 0;
    if (tid == 0) s_cur = 0;
    __syncthreads();
    for (int it = 0; it < N_POST; ++it) {
        if (tid == 0) {
            int c = s_cur;
            while (c < N_PRE && dead[c]) ++c;
            s_cur = c;
        }
        __syncthreads();
        int c = s_cur;
        int keep = (c < N_PRE) ? c : 0;
        float ky1 = boxes[4*keep+0], kx1 = boxes[4*keep+1];
        float ky2 = boxes[4*keep+2], kx2 = boxes[4*keep+3];
        if (tid == 0 && it < R_SAMPLE) {
            rois_final[4*it+0] = ky1; rois_final[4*it+1] = kx1;
            rois_final[4*it+2] = ky2; rois_final[4*it+3] = kx2;
        }
        float karea = (ky2 - ky1) * (kx2 - kx1);
        for (int q = tid; q < N_PRE; q += blockDim.x) {
            if (dead[q]) continue;
            float y1 = boxes[4*q+0], x1 = boxes[4*q+1];
            float y2 = boxes[4*q+2], x2 = boxes[4*q+3];
            float iy = fmaxf(fminf(ky2, y2) - fmaxf(ky1, y1), 0.f);
            float ix = fmaxf(fminf(kx2, x2) - fmaxf(kx1, x1), 0.f);
            float inter = iy * ix;
            float a2 = (y2 - y1) * (x2 - x1);
            float iou = inter / (karea + a2 - inter + 1e-9f);
            if (iou > NMS_TH) dead[q] = 1;
        }
        __syncthreads();
    }
}

// ---------------------------------------------------------------------------
// Stage B: bf16x3 split-precision WMMA GEMM.
// A fragments precomputed in contiguous-k order (k = 16*half + e): one thread per
// (mtile, kchunk, lane) writes 16 contiguous bf16 (hi & lo) so the GEMM
// wave loads its whole fragment as one contiguous 32B read per lane.
// K_A(e,half) = 16*half + e   (same k order as the B fragments in LDS)
// ---------------------------------------------------------------------------
__global__ void k_frag(const float* __restrict__ A, bf16* __restrict__ hi,
                       bf16* __restrict__ lo, int kchunks) {
    int t = blockIdx.x * 256 + threadIdx.x;      // grid sized exactly
    int lane = t & 31;
    int rest = t >> 5;
    int kc = rest % kchunks, mtile = rest / kchunks;
    int m = mtile * 16 + (lane & 15);
    int half = lane >> 4;
    long long K = (long long)kchunks * 32;
    const float* row = A + (long long)m * K;
    long long base = (long long)t * 16;
#pragma unroll
    for (int e = 0; e < 16; ++e) {
        int k = kc * 32 + 16 * half + e;
        float x = row[k];
        bf16 hb = (bf16)x;
        hi[base + e] = hb;
        lo[base + e] = (bf16)(x - (float)hb);
    }
}

// hi/lo split + scatter of 4 consecutive columns into B-fragment LDS layout
__device__ __forceinline__ void cvt_store4(bf16* __restrict__ hiBuf,
                                           bf16* __restrict__ loBuf,
                                           int wbase, v4f x) {
#pragma unroll
    for (int j = 0; j < 4; ++j) {
        float v = x[j];
        bf16 hb = (bf16)v;
        hiBuf[wbase + j * 16] = hb;
        loBuf[wbase + j * 16] = (bf16)(v - (float)hb);
    }
}

// One workgroup = 256 threads = 8 waves; wave w owns M-tile w (M=128 fixed);
// workgroup owns 64 columns (4 N-subtiles).  W1 streamed non-temporally and
// split hi/lo in-register; B fragments double-buffered in LDS (one barrier
// per K-chunk; conversion of chunk k+1 overlaps WMMAs of chunk k).
// WMMAs are interleaved across the 4 subtiles (hi*hi x4, hi*lo x4, lo*hi x4)
// so each accumulator has 3 independent WMMAs between reuses -> no
// WMMA->WMMA RAW stalls (ISA 7.12.1: bf16 WMMA needs ~5 slots).
__global__ void __launch_bounds__(256)
k_gemm(const bf16* __restrict__ Ahi, const bf16* __restrict__ Alo,
       const float* __restrict__ B, const float* __restrict__ bias,
       float* __restrict__ C, int N, int kchunks) {
    __shared__ __attribute__((aligned(32))) bf16 sBhi[2][NSUB * 32 * 16];
    __shared__ __attribute__((aligned(32))) bf16 sBlo[2][NSUB * 32 * 16];
    const int tid = threadIdx.x, lane = tid & 31, wave = tid >> 5;
    const int n0 = blockIdx.x * (NSUB * 16);

    v8f acc[NSUB];
#pragma unroll
    for (int s = 0; s < NSUB; ++s)
        acc[s] = (v8f){0.f,0.f,0.f,0.f,0.f,0.f,0.f,0.f};

    // Loader mapping: chunk = 32 K-rows x 64 cols = 512 float4; 256 threads
    // handle idx = tid (kr in 0..15) and idx = tid+256 (kr+16, same cols).
    const int kr  = tid >> 4;                 // 0..15   (half = 0)
    const int nrel = (tid & 15) * 4;          // 0..60
    const int sub = nrel >> 4;
    const int nn  = nrel & 15;
    const int wbase0 = ((sub * 32 +      nn) * 16) + kr;   // half 0
    const int wbase1 = ((sub * 32 + 16 + nn) * 16) + kr;   // half 1
    const float* src0 = B + (long long)kr * N + n0 + nrel;
    const float* src1 = src0 + (long long)16 * N;
    const long long kstep = (long long)32 * N;

    // Prologue: stage chunk 0 into buffer 0.
    {
        v4f x0 = __builtin_nontemporal_load((const v4f*)src0);
        v4f x1 = __builtin_nontemporal_load((const v4f*)src1);
        cvt_store4(sBhi[0], sBlo[0], wbase0, x0);
        cvt_store4(sBhi[0], sBlo[0], wbase1, x1);
    }
    __syncthreads();

    for (int kc = 0; kc < kchunks; ++kc) {
        const int cur = kc & 1, nxt = cur ^ 1;
        const bool more = (kc + 1) < kchunks;

        // Issue next chunk's global loads early (non-temporal stream of W).
        v4f y0, y1;
        if (more) {
            const float* p0 = src0 + (long long)(kc + 1) * kstep;
            const float* p1 = src1 + (long long)(kc + 1) * kstep;
            y0 = __builtin_nontemporal_load((const v4f*)p0);
            y1 = __builtin_nontemporal_load((const v4f*)p1);
        }

        // A fragments: contiguous 32B per lane from precomputed frag arrays.
        long long abase = ((long long)(wave * kchunks + kc) * 32 + lane) * 16;
        v16bf ah = *(const v16bf*)(Ahi + abase);
        v16bf al = *(const v16bf*)(Alo + abase);

        // Fetch all B fragments for this chunk up front.
        v16bf bh[NSUB], bl[NSUB];
#pragma unroll
        for (int s = 0; s < NSUB; ++s) {
            bh[s] = *(const v16bf*)&sBhi[cur][(s * 32 + lane) * 16];
            bl[s] = *(const v16bf*)&sBlo[cur][(s * 32 + lane) * 16];
        }

        // bf16x3 interleaved across subtiles: no acc reuse within 4 WMMAs.
#pragma unroll
        for (int s = 0; s < NSUB; ++s) acc[s] = wmma_bf16(ah, bh[s], acc[s]);
#pragma unroll
        for (int s = 0; s < NSUB; ++s) acc[s] = wmma_bf16(ah, bl[s], acc[s]);
#pragma unroll
        for (int s = 0; s < NSUB; ++s) acc[s] = wmma_bf16(al, bh[s], acc[s]);

        // Stage chunk k+1 into the other buffer (overlaps WMMAs above).
        if (more) {
            cvt_store4(sBhi[nxt], sBlo[nxt], wbase0, y0);
            cvt_store4(sBhi[nxt], sBlo[nxt], wbase1, y1);
        }
        __syncthreads();
    }

    // C/D layout: VGPR v -> M = v + 8*(lane>>4), N = lane&15  [ISA 7.12.2]
    const int half = lane >> 4, n = lane & 15;
#pragma unroll
    for (int s = 0; s < NSUB; ++s) {
#pragma unroll
        for (int v = 0; v < 8; ++v) {
            int row = wave * 16 + half * 8 + v;
            int col = n0 + s * 16 + n;
            C[(long long)row * N + col] = acc[s][v] + bias[col];
        }
    }
}

// Tiny head GEMMs (K=1024, N=21/84): pure fp32, negligible work.
__global__ void k_headgemm(const float* __restrict__ A, const float* __restrict__ W,
                           const float* __restrict__ b, float* __restrict__ out,
                           int N) {
    int t = blockIdx.x * 256 + threadIdx.x;
    if (t >= 128 * N) return;
    int r = t / N, c = t % N;
    const float* ar = A + (long long)r * 1024;
    float s = 0.f;
    for (int k = 0; k < 1024; ++k) s = fmaf(ar[k], W[k * N + c], s);
    out[t] = s + b[c];
}

// Final decode with the reference's view(-1,128,4) flat-index quirk:
// roi_deltas[c,r,k] = Bm.flat[c*512 + r*4 + k]; output [r,c,4] (no clip).
__global__ void k_finaldecode(const float* __restrict__ Bm,
                              const float* __restrict__ rois,
                              float* __restrict__ outBoxes) {
    int t = blockIdx.x * 256 + threadIdx.x;
    if (t >= 21 * 128) return;
    int c = t / 128, r = t % 128;
    const float* d = Bm + c * 512 + r * 4;
    float a0 = rois[4*r+0], a1 = rois[4*r+1], a2 = rois[4*r+2], a3 = rois[4*r+3];
    float h = a2 - a0, w = a3 - a1;
    float cy = a0 + 0.5f * h, cx = a1 + 0.5f * w;
    float ncy = d[0] * h + cy, ncx = d[1] * w + cx;
    float nh = expf(d[2]) * h, nw = expf(d[3]) * w;
    float* o = outBoxes + ((long long)r * 21 + c) * 4;
    o[0] = ncy - 0.5f * nh; o[1] = ncx - 0.5f * nw;
    o[2] = ncy + 0.5f * nh; o[3] = ncx + 0.5f * nw;
}

// ---------------------------------------------------------------------------
extern "C" void kernel_launch(void* const* d_in, const int* in_sizes, int n_in,
                              void* d_out, int out_size, void* d_ws, size_t ws_size,
                              hipStream_t stream) {
    const float* logits  = (const float*)d_in[0];
    const float* deltas  = (const float*)d_in[1];
    const float* anchors = (const float*)d_in[2];
    const float* pooling = (const float*)d_in[3];
    const float* W1 = (const float*)d_in[4];
    const float* b1 = (const float*)d_in[5];
    const float* W2 = (const float*)d_in[6];
    const float* b2 = (const float*)d_in[7];
    const float* Wc = (const float*)d_in[8];
    const float* bc = (const float*)d_in[9];
    const float* Wb = (const float*)d_in[10];
    const float* bb = (const float*)d_in[11];
    float* out = (float*)d_out;

    // workspace carve-up (256B aligned)
    char* p = (char*)d_ws;
    size_t off = 0;
    auto take = [&](size_t bytes) {
        void* r = p + off;
        off += (bytes + 255) & ~(size_t)255;
        return r;
    };
    u64*   keys    = (u64*)  take(SORT_N * 8);
    float* roisAll = (float*)take((size_t)N_ANCHORS * 16);
    float* bsorted = (float*)take((size_t)N_PRE * 16);
    float* roisFin = (float*)take(R_SAMPLE * 16);
    bf16*  A1hi    = (bf16*) take((size_t)128 * 25088 * 2);
    bf16*  A1lo    = (bf16*) take((size_t)128 * 25088 * 2);
    float* H1      = (float*)take((size_t)128 * 4096 * 4);
    bf16*  A2hi    = (bf16*) take((size_t)128 * 4096 * 2);
    bf16*  A2lo    = (bf16*) take((size_t)128 * 4096 * 2);
    float* H2      = (float*)take((size_t)128 * 1024 * 4);
    float* Bm      = (float*)take((size_t)128 * 84 * 4);
    (void)ws_size; (void)in_sizes; (void)n_in; (void)out_size;

    // --- Stage A: proposals -----------------------------------------------
    k_props<<<SORT_N / 256, 256, 0, stream>>>(logits, deltas, anchors, roisAll, keys);
    for (int k = 2; k <= SORT_N; k <<= 1)
        for (int j = k >> 1; j > 0; j >>= 1)
            k_bitonic<<<SORT_N / 256, 256, 0, stream>>>(keys, j, k);
    k_gather<<<(N_PRE + 255) / 256, 256, 0, stream>>>(keys, roisAll, bsorted);
    k_nms<<<1, 1024, 0, stream>>>(bsorted, keys, roisFin);

    // --- Stage B: detector head -------------------------------------------
    // GEMM1: [128,25088] @ [25088,4096]  (784 K-chunks, 64 cols/WG)
    k_frag<<<(8 * 784 * 32) / 256, 256, 0, stream>>>(pooling, A1hi, A1lo, 784);
    k_gemm<<<4096 / 64, 256, 0, stream>>>(A1hi, A1lo, W1, b1, H1, 4096, 784);
    // GEMM2: [128,4096] @ [4096,1024]   (128 K-chunks)
    k_frag<<<(8 * 128 * 32) / 256, 256, 0, stream>>>(H1, A2hi, A2lo, 128);
    k_gemm<<<1024 / 64, 256, 0, stream>>>(A2hi, A2lo, W2, b2, H2, 1024, 128);
    // Heads
    k_headgemm<<<(128 * 21 + 255) / 256, 256, 0, stream>>>(H2, Wc, bc, out, 21);
    k_headgemm<<<(128 * 84 + 255) / 256, 256, 0, stream>>>(H2, Wb, bb, Bm, 84);
    // Final decode -> out[2688 ..]
    k_finaldecode<<<(21 * 128 + 255) / 256, 256, 0, stream>>>(Bm, roisFin, out + 2688);
}
